// Head_90881507983860
// MI455X (gfx1250) — matrix operand order repaired and checked by reading, not compiled, not run on hardware
//
#include <hip/hip_runtime.h>

// Problem constants (from reference): B=16, T=2048, C=1024, H=64
#define BB 16
#define TT 2048
#define CC 1024
#define HH 64

typedef __attribute__((ext_vector_type(16))) __bf16 v16bf;
typedef __attribute__((ext_vector_type(8)))  float  v8f;

union Frag {
  v16bf v;
  uint4 u[2];
  unsigned short h[16];
};

// float -> bf16 bits, round-to-nearest-even
__device__ __forceinline__ unsigned short f2bf(float f) {
  unsigned u = __builtin_bit_cast(unsigned, f);
  u += 0x7FFFu + ((u >> 16) & 1u);
  return (unsigned short)(u >> 16);
}

__device__ __forceinline__ void cvt4(unsigned short* d, float4 f) {
  d[0] = f2bf(f.x); d[1] = f2bf(f.y); d[2] = f2bf(f.z); d[3] = f2bf(f.w);
}

// ---------------------------------------------------------------------------
// Kernel 1: pack Wq|Wk|Wv (f32, [C,H]) into WT bf16 [3][H][C] (transposed) so
// WMMA B-fragments (lane = column n, elements = two runs of 8 contiguous K) are two
// 16-byte contiguous loads.
// ---------------------------------------------------------------------------
__global__ void pack_w_kernel(const float* __restrict__ Wq,
                              const float* __restrict__ Wk,
                              const float* __restrict__ Wv,
                              unsigned short* __restrict__ WT) {
  int idx = blockIdx.x * blockDim.x + threadIdx.x;
  if (idx >= 3 * CC * HH) return;
  int m   = idx / (CC * HH);
  int rem = idx - m * (CC * HH);
  int c   = rem / HH;
  int h   = rem - c * HH;
  const float* W = (m == 0) ? Wq : (m == 1) ? Wk : Wv;
  WT[((size_t)m * HH + h) * CC + c] = f2bf(W[(size_t)c * HH + h]);
}

// ---------------------------------------------------------------------------
// Kernel 2: projections. One wave computes a 16-row tile of q,k,v (N=64 each,
// 4 N-tiles of 16) over K=C=1024 in steps of 32 -> 12 wmma per K step.
// B fragments are loaded in independent batches of 3 so the compiler can
// clause the global_load_b128s and overlap them with preceding WMMAs.
// q is pre-scaled by H^-0.5. v is written transposed: vT[B][H][T] bf16.
// ---------------------------------------------------------------------------
__global__ __launch_bounds__(256) void proj_kernel(
    const float* __restrict__ x, const unsigned short* __restrict__ WT,
    unsigned short* __restrict__ q, unsigned short* __restrict__ k,
    unsigned short* __restrict__ vT) {
  const int wave = blockIdx.x * 8 + (threadIdx.x >> 5);
  const int lane = threadIdx.x & 31;
  const int n    = lane & 15;
  const int half = lane >> 4;
  const int b    = wave >> 7;            // wave / (T/16)
  const int trow = (wave & 127) << 4;    // 16-row tile base

  // A-matrix row for this lane: M = n  (lanes 0-15 and 16-31 both hold M=0..15)
  const float* xrow = x + ((size_t)b * TT + trow + n) * CC;

  v8f acc[12] = {};  // [q0..q3, k0..k3, v0..v3]

  for (int kc = 0; kc < CC; kc += 32) {
    // A fragment: K(e) = 16*(e>>3) + 8*half + (e&7)  -> two 32B contiguous loads
    Frag a;
    const float* ap = xrow + kc + half * 8;
    cvt4(a.h + 0,  *(const float4*)(ap + 0));
    cvt4(a.h + 4,  *(const float4*)(ap + 4));
    cvt4(a.h + 8,  *(const float4*)(ap + 16));
    cvt4(a.h + 12, *(const float4*)(ap + 20));

#pragma unroll
    for (int ht = 0; ht < 4; ++ht) {
      Frag bf[3];  // independent buffers -> loads can be claused/overlapped
#pragma unroll
      for (int m = 0; m < 3; ++m) {
        const unsigned short* wp =
            WT + ((size_t)m * HH + ht * 16 + n) * CC + kc + 8 * half;
        bf[m].u[0] = *(const uint4*)(wp);
        bf[m].u[1] = *(const uint4*)(wp + 16);
      }
#pragma unroll
      for (int m = 0; m < 3; ++m) {
        acc[m * 4 + ht] = __builtin_amdgcn_wmma_f32_16x16x32_bf16(
            false, a.v, false, bf[m].v, (short)0, acc[m * 4 + ht], false, false);
      }
    }
  }

  // C/D layout: lane col = n, VGPR r -> row = 8*half + r
#pragma unroll
  for (int ht = 0; ht < 4; ++ht) {
#pragma unroll
    for (int r = 0; r < 8; ++r) {
      int t = trow + half * 8 + r;
      size_t qi = ((size_t)b * TT + t) * HH + ht * 16 + n;
      q[qi] = f2bf(acc[ht][r] * 0.125f);   // fold softmax scale H^-0.5 into q
      k[qi] = f2bf(acc[4 + ht][r]);
    }
    // vT[b][h][t]: fixed h per lane, t contiguous over r -> one 16B store
    union { unsigned short hh[8]; uint4 u; } pk;
#pragma unroll
    for (int r = 0; r < 8; ++r) pk.hh[r] = f2bf(acc[8 + ht][r]);
    *(uint4*)(vT + ((size_t)b * HH + ht * 16 + n) * TT + trow + half * 8) = pk.u;
  }
}

// ---------------------------------------------------------------------------
// Kernel 3: flash attention. One wave owns a 16-row Q band; streams K/V in
// 64-wide s-chunks with online softmax. 16 wmma per chunk (8 S + 8 PV),
// amortizing the cross-lane softmax reductions over 2x the matrix work.
// ---------------------------------------------------------------------------
__global__ __launch_bounds__(256) void attn_kernel(
    const unsigned short* __restrict__ q, const unsigned short* __restrict__ k,
    const unsigned short* __restrict__ vT, float* __restrict__ out) {
  __shared__ __attribute__((aligned(16))) unsigned short ldsP[8][16 * 64];
  const int wslot = threadIdx.x >> 5;
  const int wave  = blockIdx.x * 8 + wslot;
  const int lane  = threadIdx.x & 31;
  const int n     = lane & 15;
  const int half  = lane >> 4;
  const int b     = wave >> 7;
  const int trow  = (wave & 127) << 4;

  // Q A-fragments for K-dim = H: two frags (h 0..31, 32..63)
  Frag qa[2];
  {
    const unsigned short* qrow = q + ((size_t)b * TT + trow + n) * HH;
#pragma unroll
    for (int kk = 0; kk < 2; ++kk) {
      const unsigned short* p = qrow + kk * 32 + half * 8;
      qa[kk].u[0] = *(const uint4*)(p);
      qa[kk].u[1] = *(const uint4*)(p + 16);
    }
  }

  v8f o[4] = {};
  float mrow[8], lrow[8];
#pragma unroll
  for (int r = 0; r < 8; ++r) { mrow[r] = -__builtin_inff(); lrow[r] = 0.f; }

  const unsigned short* kbase = k  + (size_t)b * TT * HH;
  const unsigned short* vbase = vT + (size_t)b * HH * TT;
  const int nchunk = (trow + 79) >> 6;  // 64-wide s-chunks covering [0, trow+16)

  for (int c = 0; c < nchunk; ++c) {
    const int sbase = c << 6;
    if (c + 1 < nchunk)  // prefetch next chunk's K rows -> global_prefetch_b8
      __builtin_prefetch(kbase + (size_t)(sbase + 64 + n) * HH, 0, 1);

    // ---- S = Q K^T : four 16x16 tiles (cols sbase + 16j + n) ----
    v8f s[4] = {};
#pragma unroll
    for (int j = 0; j < 4; ++j) {
#pragma unroll
      for (int kk = 0; kk < 2; ++kk) {
        const unsigned short* kp =
            kbase + (size_t)(sbase + 16 * j + n) * HH + kk * 32 + half * 8;
        Frag bf;
        bf.u[0] = *(const uint4*)kp;
        bf.u[1] = *(const uint4*)(kp + 16);
        s[j] = __builtin_amdgcn_wmma_f32_16x16x32_bf16(
            false, qa[kk].v, false, bf.v, (short)0, s[j], false, false);
      }
    }

    // ---- causal mask (only needed near the diagonal) ----
    if (sbase + 63 > trow) {
#pragma unroll
      for (int j = 0; j < 4; ++j) {
#pragma unroll
        for (int r = 0; r < 8; ++r) {
          int t = trow + half * 8 + r;
          if (sbase + 16 * j + n > t) s[j][r] = -__builtin_inff();
        }
      }
    }

    // ---- online softmax; overwrite s[] with P in place ----
#pragma unroll
    for (int r = 0; r < 8; ++r) {
      float mx = fmaxf(fmaxf(s[0][r], s[1][r]), fmaxf(s[2][r], s[3][r]));
#pragma unroll
      for (int off = 1; off < 16; off <<= 1)
        mx = fmaxf(mx, __shfl_xor(mx, off, 32));
      float mnew  = fmaxf(mrow[r], mx);
      float alpha = __expf(mrow[r] - mnew);
      mrow[r] = mnew;
      s[0][r] = __expf(s[0][r] - mnew);
      s[1][r] = __expf(s[1][r] - mnew);
      s[2][r] = __expf(s[2][r] - mnew);
      s[3][r] = __expf(s[3][r] - mnew);
      float rs = (s[0][r] + s[1][r]) + (s[2][r] + s[3][r]);
#pragma unroll
      for (int off = 1; off < 16; off <<= 1)
        rs += __shfl_xor(rs, off, 32);
      lrow[r] = lrow[r] * alpha + rs;
      o[0][r] *= alpha; o[1][r] *= alpha; o[2][r] *= alpha; o[3][r] *= alpha;
    }

    // ---- P: C-layout regs -> LDS -> A-fragment relayout ----
    unsigned short* pl = &ldsP[wslot][0];
#pragma unroll
    for (int j = 0; j < 4; ++j) {
#pragma unroll
      for (int r = 0; r < 8; ++r) {
        int row = half * 8 + r;
        pl[row * 64 + 16 * j + n] = f2bf(s[j][r]);
      }
    }
    asm volatile("s_wait_dscnt 0" ::: "memory");

    // ---- O += P * V : A = P (16x64 in 2 frags), B from vT (contiguous) ----
#pragma unroll
    for (int kk = 0; kk < 2; ++kk) {
      Frag pa;
      const unsigned short* pr = pl + n * 64 + kk * 32 + half * 8;  // row M = n
      pa.u[0] = *(const uint4*)(pr);
      pa.u[1] = *(const uint4*)(pr + 16);
#pragma unroll
      for (int ht = 0; ht < 4; ++ht) {
        const unsigned short* vp =
            vbase + (size_t)(ht * 16 + n) * TT + sbase + kk * 32 + half * 8;
        Frag vb;
        vb.u[0] = *(const uint4*)vp;
        vb.u[1] = *(const uint4*)(vp + 16);
        o[ht] = __builtin_amdgcn_wmma_f32_16x16x32_bf16(
            false, pa.v, false, vb.v, (short)0, o[ht], false, false);
      }
    }
  }

  // ---- epilogue: normalize and store fp32 ----
#pragma unroll
  for (int ht = 0; ht < 4; ++ht) {
#pragma unroll
    for (int r = 0; r < 8; ++r) {
      int t = trow + half * 8 + r;
      out[((size_t)b * TT + t) * HH + ht * 16 + n] = o[ht][r] / lrow[r];
    }
  }
}

// ---------------------------------------------------------------------------
// Workspace layout (bytes):
//   [0x000000) WT  bf16 [3][64][1024]  = 384 KiB
//   [1 MiB)    q   bf16 [16][2048][64] = 4 MiB
//   [5 MiB)    k   bf16 [16][2048][64] = 4 MiB
//   [9 MiB)    vT  bf16 [16][64][2048] = 4 MiB     (total 13 MiB)
// ---------------------------------------------------------------------------
extern "C" void kernel_launch(void* const* d_in, const int* in_sizes, int n_in,
                              void* d_out, int out_size, void* d_ws, size_t ws_size,
                              hipStream_t stream) {
  const float* x  = (const float*)d_in[0];
  const float* Wq = (const float*)d_in[1];
  const float* Wk = (const float*)d_in[2];
  const float* Wv = (const float*)d_in[3];
  float* out = (float*)d_out;

  char* ws = (char*)d_ws;
  unsigned short* WT = (unsigned short*)(ws);
  unsigned short* qb = (unsigned short*)(ws + (size_t)(1) * 1024 * 1024);
  unsigned short* kb = (unsigned short*)(ws + (size_t)(5) * 1024 * 1024);
  unsigned short* vb = (unsigned short*)(ws + (size_t)(9) * 1024 * 1024);

  // 1) pack weights (bf16, transposed)
  pack_w_kernel<<<(3 * CC * HH + 255) / 256, 256, 0, stream>>>(Wq, Wk, Wv, WT);

  // 2) q/k/vT projections: 2048 row tiles, 1 wave each, 8 waves per block
  proj_kernel<<<256, 256, 0, stream>>>(x, WT, qb, kb, vb);

  // 3) flash attention: same wave mapping
  attn_kernel<<<256, 256, 0, stream>>>(qb, kb, vb, out);
}
